// CentroidModel_7490422964419
// MI455X (gfx1250) — matrix operand order repaired and checked by reading, not compiled, not run on hardware
//
#include <hip/hip_runtime.h>

typedef __attribute__((ext_vector_type(2)))  float v2f;
typedef __attribute__((ext_vector_type(4)))  float v4f;
typedef __attribute__((ext_vector_type(8)))  float v8f;
typedef __attribute__((ext_vector_type(4)))  unsigned int v4u;
typedef __attribute__((ext_vector_type(8)))  __bf16 v8bf;
typedef __attribute__((ext_vector_type(16))) __bf16 v16bf;

#define NB    8192   // batch rows
#define DIN   1024
#define DMID  768
#define NC    16384  // centroids
#define LDSS  20     // fp32 LDS row stride (floats): 80B, b64-aligned, bank-friendly
#define LDSB  40     // bf16 LDS row stride (elements): 80B, b128-aligned

// ---------------------------------------------------------------------------
// GEMM1: e[NB,DMID] = x[NB,DIN] @ W[DIN,DMID], exact fp32 via V_WMMA_F32_16X16X4_F32
// (6% of total FLOPs; kept full precision since it feeds e and e_sq)
// Workgroup: 256 thr = 8 waves, tile 128(M) x 128(N). Wave: 64x32 = 4x2 16x16 tiles.
// ---------------------------------------------------------------------------
__global__ __launch_bounds__(256) void gemm_xw_kernel(const float* __restrict__ x,
                                                      const float* __restrict__ W,
                                                      float* __restrict__ e) {
  __shared__ float lds_a[128 * LDSS];   // x tile:  [m][k]
  __shared__ float lds_b[128 * LDSS];   // W tile transposed: [n][k]

  const int tid  = threadIdx.x;
  const int lane = tid & 31;
  const int l16  = lane & 15;
  const int kh   = (lane >> 4) * 2;     // K sub-offset per half-wave (f32 A/B frags)
  const int wave = tid >> 5;
  const int wm   = (wave & 1) * 64;
  const int wn   = (wave >> 1) * 32;
  const int m_blk = blockIdx.y * 128;
  const int n_blk = blockIdx.x * 128;

  v8f acc[4][2];
#pragma unroll
  for (int i = 0; i < 4; ++i)
#pragma unroll
    for (int j = 0; j < 2; ++j) acc[i][j] = (v8f)0.0f;

  const int arow = tid >> 1;            // 0..127
  const int acol = (tid & 1) * 8;       // 0 or 8
  const int bk   = tid >> 4;            // 0..15
  const int bn   = (tid & 15) * 8;      // 0..120

  for (int kb = 0; kb < DIN; kb += 16) {
    const float* gx = x + (size_t)(m_blk + arow) * DIN + kb + acol;
    v4f a0 = *(const v4f*)gx;
    v4f a1 = *(const v4f*)(gx + 4);
    *(v4f*)&lds_a[arow * LDSS + acol]     = a0;
    *(v4f*)&lds_a[arow * LDSS + acol + 4] = a1;
    const float* gw = W + (size_t)(kb + bk) * DMID + n_blk + bn;
    v4f b0 = *(const v4f*)gw;
    v4f b1 = *(const v4f*)(gw + 4);
#pragma unroll
    for (int i = 0; i < 4; ++i) lds_b[(bn + i) * LDSS + bk]     = b0[i];
#pragma unroll
    for (int i = 0; i < 4; ++i) lds_b[(bn + 4 + i) * LDSS + bk] = b1[i];
    __syncthreads();

#pragma unroll
    for (int ks = 0; ks < 4; ++ks) {
      const int kc = ks * 4 + kh;
      v2f af[4], bf[2];
#pragma unroll
      for (int mt = 0; mt < 4; ++mt)
        af[mt] = *(const v2f*)&lds_a[(wm + mt * 16 + l16) * LDSS + kc];
#pragma unroll
      for (int nt = 0; nt < 2; ++nt)
        bf[nt] = *(const v2f*)&lds_b[(wn + nt * 16 + l16) * LDSS + kc];
#pragma unroll
      for (int mt = 0; mt < 4; ++mt)
#pragma unroll
        for (int nt = 0; nt < 2; ++nt)
          acc[mt][nt] = __builtin_amdgcn_wmma_f32_16x16x4_f32(
              false, af[mt], false, bf[nt], (short)0, acc[mt][nt], false, false);
    }
    __syncthreads();
  }

  const int mrow0 = m_blk + wm + 8 * (lane >> 4);
  const int ncol0 = n_blk + wn + l16;
#pragma unroll
  for (int mt = 0; mt < 4; ++mt)
#pragma unroll
    for (int nt = 0; nt < 2; ++nt)
#pragma unroll
      for (int r = 0; r < 8; ++r)
        e[(size_t)(mrow0 + mt * 16 + r) * DMID + ncol0 + nt * 16] = acc[mt][nt][r];
}

// ---------------------------------------------------------------------------
// Row squared-norm: one wave32 per row.
// ---------------------------------------------------------------------------
__global__ __launch_bounds__(256) void rownorm_kernel(const float* __restrict__ mat,
                                                      float* __restrict__ out,
                                                      int ncols) {
  const int row  = blockIdx.x * 8 + (threadIdx.x >> 5);
  const int lane = threadIdx.x & 31;
  const float* p = mat + (size_t)row * ncols;
  float s = 0.0f;
  for (int k = lane * 4; k < ncols; k += 128) {
    v4f v = *(const v4f*)(p + k);
    s += v.x * v.x + v.y * v.y + v.z * v.z + v.w * v.w;
  }
#pragma unroll
  for (int off = 16; off > 0; off >>= 1) s += __shfl_xor(s, off, 32);
  if (lane == 0) out[row] = s;
}

// ---------------------------------------------------------------------------
// bf16 hi/lo split helper (truncation split: exact to ~2^-17 relative)
// ---------------------------------------------------------------------------
__device__ __forceinline__ void split2(float f0, float f1,
                                       unsigned int& h, unsigned int& l) {
  unsigned int b0 = __float_as_uint(f0);
  unsigned int b1 = __float_as_uint(f1);
  float r0 = f0 - __uint_as_float(b0 & 0xffff0000u);
  float r1 = f1 - __uint_as_float(b1 & 0xffff0000u);
  h = (b0 >> 16) | (b1 & 0xffff0000u);
  l = (__float_as_uint(r0) >> 16) | (__float_as_uint(r1) & 0xffff0000u);
}

__device__ __forceinline__ v16bf cat8(v8bf a, v8bf b) {
  return __builtin_shufflevector(a, b, 0, 1, 2, 3, 4, 5, 6, 7,
                                        8, 9, 10, 11, 12, 13, 14, 15);
}

// ---------------------------------------------------------------------------
// GEMM2 + epilogue: out[b][c] = 2*(e @ centroids^T) - e_sq[b] - c_sq[c]
// bf16x3 split: a*b ~= ah*bh + ah*bl + al*bh via V_WMMA_F32_16X16X32_BF16.
// 3 bf16 WMMAs replace 8 fp32 WMMAs per 16x16x32 -> 2.7x fewer matrix ops.
// K-block = 32. LDS holds hi/lo bf16 tiles for both operands (40 KB).
// ---------------------------------------------------------------------------
__global__ __launch_bounds__(256) void gemm_ec_kernel(const float* __restrict__ e,
                                                      const float* __restrict__ cent,
                                                      const float* __restrict__ e_sq,
                                                      const float* __restrict__ c_sq,
                                                      float* __restrict__ out) {
  // packed bf16 tiles, stored as uint pairs; row stride LDSB bf16 = LDSB/2 uints
  __shared__ unsigned int lds_ah[128 * (LDSB / 2)];
  __shared__ unsigned int lds_al[128 * (LDSB / 2)];
  __shared__ unsigned int lds_bh[128 * (LDSB / 2)];
  __shared__ unsigned int lds_bl[128 * (LDSB / 2)];

  const int tid  = threadIdx.x;
  const int lane = tid & 31;
  const int l16  = lane & 15;
  const int hw   = lane >> 4;           // half-wave index
  const int wave = tid >> 5;
  const int wm   = (wave & 1) * 64;
  const int wn   = (wave >> 1) * 32;
  const int m_blk = blockIdx.y * 128;
  const int n_blk = blockIdx.x * 128;

  // fragment K offsets per ISA 7.12.2 (16-bit operands, wave32):
  // A (16x32): lane holds K=[8*hw, 8*hw+8) and [16+8*hw, 16+8*hw+8)
  // B (32x16): lane holds K=[8*hw, 8*hw+8) and [16+8*hw, 16+8*hw+8)
  const int a_c0 = 8 * hw;
  const int b_c0 = 8 * hw;

  v8f acc[4][2];
#pragma unroll
  for (int i = 0; i < 4; ++i)
#pragma unroll
    for (int j = 0; j < 2; ++j) acc[i][j] = (v8f)0.0f;

  // loader: 128 rows x 32 cols fp32 per matrix, 256 thr -> 16 floats each
  const int lrow = tid >> 1;            // 0..127
  const int lcol = (tid & 1) * 16;      // 0 or 16 (fp32 col)
  const int ucol = lcol >> 1;           // 0 or 8  (uint col in packed LDS)
  const int abase = lrow * (LDSB / 2) + ucol;

  for (int kb = 0; kb < DMID; kb += 32) {
    const float* ge = e    + (size_t)(m_blk + lrow) * DMID + kb + lcol;
    const float* gc = cent + (size_t)(n_blk + lrow) * DMID + kb + lcol;
    v4f fa0 = *(const v4f*)ge;
    v4f fa1 = *(const v4f*)(ge + 4);
    v4f fa2 = *(const v4f*)(ge + 8);
    v4f fa3 = *(const v4f*)(ge + 12);
    v4f fb0 = *(const v4f*)gc;
    v4f fb1 = *(const v4f*)(gc + 4);
    v4f fb2 = *(const v4f*)(gc + 8);
    v4f fb3 = *(const v4f*)(gc + 12);

    unsigned int h[8], l[8];
    // A (e) tile split
    split2(fa0.x, fa0.y, h[0], l[0]); split2(fa0.z, fa0.w, h[1], l[1]);
    split2(fa1.x, fa1.y, h[2], l[2]); split2(fa1.z, fa1.w, h[3], l[3]);
    split2(fa2.x, fa2.y, h[4], l[4]); split2(fa2.z, fa2.w, h[5], l[5]);
    split2(fa3.x, fa3.y, h[6], l[6]); split2(fa3.z, fa3.w, h[7], l[7]);
    {
      v4u H0 = {h[0], h[1], h[2], h[3]};
      v4u H1 = {h[4], h[5], h[6], h[7]};
      v4u L0 = {l[0], l[1], l[2], l[3]};
      v4u L1 = {l[4], l[5], l[6], l[7]};
      *(v4u*)&lds_ah[abase]     = H0;
      *(v4u*)&lds_ah[abase + 4] = H1;
      *(v4u*)&lds_al[abase]     = L0;
      *(v4u*)&lds_al[abase + 4] = L1;
    }
    // B (centroids) tile split
    split2(fb0.x, fb0.y, h[0], l[0]); split2(fb0.z, fb0.w, h[1], l[1]);
    split2(fb1.x, fb1.y, h[2], l[2]); split2(fb1.z, fb1.w, h[3], l[3]);
    split2(fb2.x, fb2.y, h[4], l[4]); split2(fb2.z, fb2.w, h[5], l[5]);
    split2(fb3.x, fb3.y, h[6], l[6]); split2(fb3.z, fb3.w, h[7], l[7]);
    {
      v4u H0 = {h[0], h[1], h[2], h[3]};
      v4u H1 = {h[4], h[5], h[6], h[7]};
      v4u L0 = {l[0], l[1], l[2], l[3]};
      v4u L1 = {l[4], l[5], l[6], l[7]};
      *(v4u*)&lds_bh[abase]     = H0;
      *(v4u*)&lds_bh[abase + 4] = H1;
      *(v4u*)&lds_bl[abase]     = L0;
      *(v4u*)&lds_bl[abase + 4] = L1;
    }
    __syncthreads();

    const __bf16* pah = (const __bf16*)lds_ah;
    const __bf16* pal = (const __bf16*)lds_al;
    const __bf16* pbh = (const __bf16*)lds_bh;
    const __bf16* pbl = (const __bf16*)lds_bl;

    v16bf ah[4], al[4], bh[2], bl[2];
#pragma unroll
    for (int mt = 0; mt < 4; ++mt) {
      const int base = (wm + mt * 16 + l16) * LDSB;
      ah[mt] = cat8(*(const v8bf*)&pah[base + a_c0],
                    *(const v8bf*)&pah[base + a_c0 + 16]);
      al[mt] = cat8(*(const v8bf*)&pal[base + a_c0],
                    *(const v8bf*)&pal[base + a_c0 + 16]);
    }
#pragma unroll
    for (int nt = 0; nt < 2; ++nt) {
      const int base = (wn + nt * 16 + l16) * LDSB;
      bh[nt] = cat8(*(const v8bf*)&pbh[base + b_c0],
                    *(const v8bf*)&pbh[base + b_c0 + 16]);
      bl[nt] = cat8(*(const v8bf*)&pbl[base + b_c0],
                    *(const v8bf*)&pbl[base + b_c0 + 16]);
    }

#pragma unroll
    for (int mt = 0; mt < 4; ++mt)
#pragma unroll
      for (int nt = 0; nt < 2; ++nt) {
        acc[mt][nt] = __builtin_amdgcn_wmma_f32_16x16x32_bf16(
            false, ah[mt], false, bh[nt], (short)0, acc[mt][nt], false, false);
        acc[mt][nt] = __builtin_amdgcn_wmma_f32_16x16x32_bf16(
            false, ah[mt], false, bl[nt], (short)0, acc[mt][nt], false, false);
        acc[mt][nt] = __builtin_amdgcn_wmma_f32_16x16x32_bf16(
            false, al[mt], false, bh[nt], (short)0, acc[mt][nt], false, false);
      }
    __syncthreads();
  }

  // epilogue: out = 2*cross - e_sq - c_sq
  const int mrow0 = m_blk + wm + 8 * hw;
  const int ncol0 = n_blk + wn + l16;
#pragma unroll
  for (int nt = 0; nt < 2; ++nt) {
    const float cs = c_sq[ncol0 + nt * 16];
#pragma unroll
    for (int mt = 0; mt < 4; ++mt) {
#pragma unroll
      for (int r = 0; r < 8; ++r) {
        const int m = mrow0 + mt * 16 + r;
        out[(size_t)m * NC + ncol0 + nt * 16] =
            2.0f * acc[mt][nt][r] - e_sq[m] - cs;
      }
    }
  }
}

// ---------------------------------------------------------------------------
extern "C" void kernel_launch(void* const* d_in, const int* in_sizes, int n_in,
                              void* d_out, int out_size, void* d_ws, size_t ws_size,
                              hipStream_t stream) {
  const float* x    = (const float*)d_in[0];
  const float* W    = (const float*)d_in[1];
  const float* cent = (const float*)d_in[2];
  float* out = (float*)d_out;

  float* e    = (float*)d_ws;                       // NB*DMID floats
  float* e_sq = e + (size_t)NB * DMID;              // NB floats
  float* c_sq = e_sq + NB;                          // NC floats

  dim3 g1(DMID / 128, NB / 128);                    // 6 x 64
  gemm_xw_kernel<<<g1, 256, 0, stream>>>(x, W, e);

  rownorm_kernel<<<NB / 8, 256, 0, stream>>>(e, e_sq, DMID);
  rownorm_kernel<<<NC / 8, 256, 0, stream>>>(cent, c_sq, DMID);

  dim3 g2(NC / 128, NB / 128);                      // 128 x 64
  gemm_ec_kernel<<<g2, 256, 0, stream>>>(e, cent, e_sq, c_sq, out);
}
